// ResamplerBlock_43877385896268
// MI455X (gfx1250) — compile-verified, matrix operand order checked by reading, not run on hardware
//
#include <hip/hip_runtime.h>
#include <hip/hip_bf16.h>
#include <math.h>

// ---------------------------------------------------------------------------
// Problem constants (fixed-shape reference)
// ---------------------------------------------------------------------------
#define BATCH   16
#define NQ      256
#define NKV     2304
#define H_DIM   768
#define IMG_DIM 1024
#define HEADS   12
#define HEAD_DIM 64
#define INTER   3072
#define QK_SCALE 0.125f   // 1/sqrt(64)
#define LN_EPS   1e-5f

typedef __attribute__((ext_vector_type(16))) __bf16 v16bf;
typedef __attribute__((ext_vector_type(8)))  float  v8f;

// ---------------------------------------------------------------------------
// WMMA fragment helpers.
//
// 16-bit A-matrix 16x32 layout (ISA 7.12.2): lanes 0-15 hold row M=lane with
// K = {0..7, 16..23}; lanes 16-31 hold row M=lane-16 with K = {8..15, 24..31}.
// So per lane: two contiguous 16-byte chunks at k0+((lane&16)?8:0) and +16.
// The B operand, fetched from a row-major B^T [N][K] buffer, uses the
// identical pattern with row = N = lane&15.
// ---------------------------------------------------------------------------
__device__ inline v16bf load_frag(const __bf16* rowptr, int k0, int lane) {
    int koff = k0 + ((lane & 16) ? 8 : 0);
    union { v16bf v; uint4 u[2]; } r;
    r.u[0] = *reinterpret_cast<const uint4*>(rowptr + koff);
    r.u[1] = *reinterpret_cast<const uint4*>(rowptr + koff + 16);
    return r.v;
}

__device__ inline v8f vzero8() {
    v8f z;
#pragma unroll
    for (int i = 0; i < 8; ++i) z[i] = 0.f;
    return z;
}

__device__ inline v8f wmma_bf16(v16bf a, v16bf b, v8f c) {
    // D = A(16x32) * B(32x16) + C, f32 accumulate
    return __builtin_amdgcn_wmma_f32_16x16x32_bf16(
        /*neg_a=*/false, a, /*neg_b=*/false, b,
        /*c_mod=*/(short)0, c, /*reuse_a=*/false, /*reuse_b=*/false);
}

// ---------------------------------------------------------------------------
// Weight convert + transpose:  Wt[n][k] = bf16(W[k][n])
// ---------------------------------------------------------------------------
__global__ __launch_bounds__(256)
void transpose_convert_kernel(const float* __restrict__ W, __bf16* __restrict__ Wt,
                              int K, int N) {
    int idx = blockIdx.x * 256 + threadIdx.x;
    if (idx >= N * K) return;
    int n = idx / K;
    int k = idx - n * K;
    Wt[idx] = (__bf16)W[(size_t)k * N + n];
}

// ---------------------------------------------------------------------------
// LayerNorm (fp32 in -> bf16 out), one row per 256-thread block (8 waves)
// ---------------------------------------------------------------------------
__global__ __launch_bounds__(256)
void ln_kernel(const float* __restrict__ X, const float* __restrict__ g,
               const float* __restrict__ beta, __bf16* __restrict__ Y, int D) {
    const int row = blockIdx.x;
    const float* xr = X + (size_t)row * D;
    float s = 0.f, s2 = 0.f;
    for (int i = threadIdx.x; i < D; i += 256) {
        float v = xr[i];
        s += v; s2 += v * v;
    }
#pragma unroll
    for (int m = 16; m >= 1; m >>= 1) {
        s  += __shfl_xor(s,  m, 32);
        s2 += __shfl_xor(s2, m, 32);
    }
    __shared__ float red[2][8];
    int w = threadIdx.x >> 5;
    if ((threadIdx.x & 31) == 0) { red[0][w] = s; red[1][w] = s2; }
    __syncthreads();
    float ts = 0.f, ts2 = 0.f;
#pragma unroll
    for (int i = 0; i < 8; ++i) { ts += red[0][i]; ts2 += red[1][i]; }
    float mu  = ts / (float)D;
    float var = ts2 / (float)D - mu * mu;
    float inv = rsqrtf(var + LN_EPS);
    __bf16* yr = Y + (size_t)row * D;
    for (int i = threadIdx.x; i < D; i += 256)
        yr[i] = (__bf16)((xr[i] - mu) * inv * g[i] + beta[i]);
}

// ---------------------------------------------------------------------------
// Generic bf16 WMMA GEMM:  C[M][N] = epilogue( A[M][K] * Wt[N][K]^T )
// Block: 256 threads = 8 waves (4 down M x 2 across N),
// block tile 128(M) x 128(N), wave tile 32x64 = 2x4 WMMA accumulators.
// Per k-step: 6 fragment loads feed 8 WMMAs. This is the sweet spot on
// wave32/gfx1250: the 64x64 variant (16 WMMAs, ~210 VGPRs) spilled
// accumulators to scratch and inserted WMMA hazard NOPs, while 32x32 needs
// 1 KB fetched per WMMA vs 768 B here. ~130 VGPRs leaves headroom for the
// compiler's load double-buffering registers (verified: no scratch, 5-deep
// staggered s_wait_loadcnt interleave).
// out_mode 0: row-major store (bf16 and/or fp32, optional residual/GELU)
// out_mode 1: head-split  outB[b][h][r][d]      (Q, K)
// out_mode 2: head-split transposed outB[b][h][d][r]  (V^T for the PV WMMA)
// ---------------------------------------------------------------------------
__global__ __launch_bounds__(256)
void gemm_bf16_kernel(const __bf16* __restrict__ A, const __bf16* __restrict__ Wt,
                      const float* __restrict__ bias,
                      __bf16* __restrict__ outB, float* __restrict__ outF,
                      const float* __restrict__ residual,
                      int M, int N, int K, float scale, int do_gelu,
                      int out_mode, int rows_per_batch) {
    const int lane  = threadIdx.x & 31;
    const int wave  = threadIdx.x >> 5;
    const int waveM = wave & 3;     // 4 waves down M
    const int waveN = wave >> 2;    // 2 waves across N
    const int col16 = lane & 15;
    const int rowBase = blockIdx.x * 128 + waveM * 32;
    const int colBase = blockIdx.y * 128 + waveN * 64;

    const __bf16* arow[2];
    const __bf16* brow[4];
#pragma unroll
    for (int s = 0; s < 2; ++s)
        arow[s] = A + (size_t)(rowBase + s * 16 + col16) * K;
#pragma unroll
    for (int s = 0; s < 4; ++s)
        brow[s] = Wt + (size_t)(colBase + s * 16 + col16) * K;

    // Software prefetch (global_prefetch_b8): pull the back half of each
    // fragment row toward L2/L0 while the front half is being consumed.
    const int kh = (K >> 1);
#pragma unroll
    for (int s = 0; s < 2; ++s) __builtin_prefetch(arow[s] + kh, 0, 3);
#pragma unroll
    for (int s = 0; s < 4; ++s) __builtin_prefetch(brow[s] + kh, 0, 3);

    v8f acc[2][4];
#pragma unroll
    for (int ms = 0; ms < 2; ++ms)
#pragma unroll
        for (int ns = 0; ns < 4; ++ns) acc[ms][ns] = vzero8();

    for (int k0 = 0; k0 < K; k0 += 32) {
        v16bf a0 = load_frag(arow[0], k0, lane);
        v16bf a1 = load_frag(arow[1], k0, lane);
        v16bf b0 = load_frag(brow[0], k0, lane);
        v16bf b1 = load_frag(brow[1], k0, lane);
        v16bf b2 = load_frag(brow[2], k0, lane);
        v16bf b3 = load_frag(brow[3], k0, lane);
        acc[0][0] = wmma_bf16(a0, b0, acc[0][0]);
        acc[0][1] = wmma_bf16(a0, b1, acc[0][1]);
        acc[0][2] = wmma_bf16(a0, b2, acc[0][2]);
        acc[0][3] = wmma_bf16(a0, b3, acc[0][3]);
        acc[1][0] = wmma_bf16(a1, b0, acc[1][0]);
        acc[1][1] = wmma_bf16(a1, b1, acc[1][1]);
        acc[1][2] = wmma_bf16(a1, b2, acc[1][2]);
        acc[1][3] = wmma_bf16(a1, b3, acc[1][3]);
    }

    // C/D layout: VGPR r holds (M = r + (lane>=16 ? 8 : 0), N = lane&15)
    const int rbase = (lane & 16) ? 8 : 0;
#pragma unroll
    for (int ms = 0; ms < 2; ++ms) {
#pragma unroll
        for (int ns = 0; ns < 4; ++ns) {
#pragma unroll
            for (int r = 0; r < 8; ++r) {
                int row = rowBase + ms * 16 + rbase + r;
                int cl  = colBase + ns * 16 + col16;
                float v = acc[ms][ns][r] + bias[cl];
                v *= scale;
                if (do_gelu) v = 0.5f * v * (1.f + erff(v * 0.70710678118f));
                if (residual) v += residual[(size_t)row * N + cl];
                if (out_mode == 0) {
                    if (outF) outF[(size_t)row * N + cl] = v;
                    if (outB) outB[(size_t)row * N + cl] = (__bf16)v;
                } else {
                    int bidx = row / rows_per_batch;
                    int rr   = row - bidx * rows_per_batch;
                    int h = cl >> 6, d = cl & 63;
                    if (out_mode == 1)
                        outB[(((size_t)bidx * HEADS + h) * rows_per_batch + rr) * HEAD_DIM + d] = (__bf16)v;
                    else
                        outB[(((size_t)bidx * HEADS + h) * HEAD_DIM + d) * (size_t)rows_per_batch + rr] = (__bf16)v;
                }
            }
        }
    }
}

// ---------------------------------------------------------------------------
// Flash attention: one wave per (b, h, 16-query tile); stream 32 keys/chunk.
// Q pre-scaled bf16 [b][h][q][64]; K bf16 [b][h][key][64] (== B^T rows for S);
// Vt bf16 [b][h][d][key]        (== B^T rows for O += P*V).
// P transposed D-layout -> A-layout through per-wave LDS (1 KB each).
// ---------------------------------------------------------------------------
__global__ __launch_bounds__(128)
void attention_kernel(const __bf16* __restrict__ Q, const __bf16* __restrict__ Kh,
                      const __bf16* __restrict__ Vt, __bf16* __restrict__ Ob) {
    __shared__ __align__(16) __bf16 ldsP[4][16 * 32];
    const int lane  = threadIdx.x & 31;
    const int wave  = threadIdx.x >> 5;
    const int task  = blockIdx.x * 4 + wave;   // 3072 tasks total
    const int qt    = task & 15;               // query tile within (b,h)
    const int bh    = task >> 4;               // 0..191
    const int col16 = lane & 15;
    const int rbase = (lane & 16) ? 8 : 0;

    // Q tile A-fragments (16 queries x 64 d), resident for the whole loop
    const __bf16* qrow = Q + ((size_t)bh * NQ + qt * 16 + col16) * HEAD_DIM;
    v16bf aq0 = load_frag(qrow, 0, lane);
    v16bf aq1 = load_frag(qrow, 32, lane);

    const __bf16* kbase = Kh + (size_t)bh * NKV * HEAD_DIM;
    const __bf16* vrow[4];
#pragma unroll
    for (int t = 0; t < 4; ++t)
        vrow[t] = Vt + ((size_t)bh * HEAD_DIM + t * 16 + col16) * NKV;

    v8f o[4];
    float m_run[8], l_run[8];
#pragma unroll
    for (int t = 0; t < 4; ++t) o[t] = vzero8();
#pragma unroll
    for (int r = 0; r < 8; ++r) { m_run[r] = -INFINITY; l_run[r] = 0.f; }

    __bf16* pw = ldsP[wave];

    for (int key0 = 0; key0 < NKV; key0 += 32) {
        // Prefetch next chunk's K rows and V spans (global_prefetch_b8)
        if (key0 + 32 < NKV) {
            __builtin_prefetch(kbase + (size_t)(key0 + 32 + col16) * HEAD_DIM, 0, 3);
            __builtin_prefetch(kbase + (size_t)(key0 + 48 + col16) * HEAD_DIM, 0, 3);
#pragma unroll
            for (int t = 0; t < 4; ++t)
                __builtin_prefetch(vrow[t] + key0 + 32, 0, 3);
        }

        // S = Q * K^T  (two 16-key column tiles, d reduced in two 32-steps)
        v8f s0 = vzero8(), s1 = vzero8();
        {
            const __bf16* kr0 = kbase + (size_t)(key0 + col16) * HEAD_DIM;
            const __bf16* kr1 = kbase + (size_t)(key0 + 16 + col16) * HEAD_DIM;
            v16bf b00 = load_frag(kr0, 0, lane), b01 = load_frag(kr0, 32, lane);
            v16bf b10 = load_frag(kr1, 0, lane), b11 = load_frag(kr1, 32, lane);
            s0 = wmma_bf16(aq0, b00, s0);
            s0 = wmma_bf16(aq1, b01, s0);
            s1 = wmma_bf16(aq0, b10, s1);
            s1 = wmma_bf16(aq1, b11, s1);
        }

        // Online softmax: rows live across the 16 lanes of each half-wave.
        float p0[8], p1[8], sc[8];
#pragma unroll
        for (int r = 0; r < 8; ++r) {
            float mx = fmaxf(s0[r], s1[r]);
            mx = fmaxf(mx, __shfl_xor(mx, 1, 16));
            mx = fmaxf(mx, __shfl_xor(mx, 2, 16));
            mx = fmaxf(mx, __shfl_xor(mx, 4, 16));
            mx = fmaxf(mx, __shfl_xor(mx, 8, 16));
            float mnew = fmaxf(m_run[r], mx);
            sc[r] = __expf(m_run[r] - mnew);
            p0[r] = __expf(s0[r] - mnew);
            p1[r] = __expf(s1[r] - mnew);
            float rs = p0[r] + p1[r];
            rs += __shfl_xor(rs, 1, 16);
            rs += __shfl_xor(rs, 2, 16);
            rs += __shfl_xor(rs, 4, 16);
            rs += __shfl_xor(rs, 8, 16);
            l_run[r] = l_run[r] * sc[r] + rs;
            m_run[r] = mnew;
        }

        // Transpose P (D-layout: row r+rbase, col=lane&15) to A-layout via LDS
#pragma unroll
        for (int r = 0; r < 8; ++r) {
            pw[(rbase + r) * 32 + col16]      = (__bf16)p0[r];
            pw[(rbase + r) * 32 + 16 + col16] = (__bf16)p1[r];
        }
        asm volatile("s_wait_dscnt 0" ::: "memory");  // same-wave LDS RAW
        v16bf pa = load_frag(pw + col16 * 32, 0, lane);

        // Rescale running O, then O += P(16x32) * V(32keys x 64d)
#pragma unroll
        for (int t = 0; t < 4; ++t)
#pragma unroll
            for (int r = 0; r < 8; ++r) o[t][r] *= sc[r];
#pragma unroll
        for (int t = 0; t < 4; ++t) {
            v16bf vb = load_frag(vrow[t] + key0, 0, lane);
            o[t] = wmma_bf16(pa, vb, o[t]);
        }
    }

    // Normalize and merge heads: attn_out[b*NQ + q][h*64 + d] (bf16)
    const int b = bh / HEADS, h = bh - b * HEADS;
#pragma unroll
    for (int t = 0; t < 4; ++t)
#pragma unroll
        for (int r = 0; r < 8; ++r) {
            float val = o[t][r] / l_run[r];
            int qr = qt * 16 + rbase + r;
            Ob[((size_t)b * NQ + qr) * H_DIM + h * HEAD_DIM + t * 16 + col16] = (__bf16)val;
        }
}

// ---------------------------------------------------------------------------
// Host-side orchestration
// ---------------------------------------------------------------------------
extern "C" void kernel_launch(void* const* d_in, const int* in_sizes, int n_in,
                              void* d_out, int out_size, void* d_ws, size_t ws_size,
                              hipStream_t stream) {
    (void)in_sizes; (void)n_in; (void)out_size; (void)ws_size;
    const float* hidden = (const float*)d_in[0];
    const float* x      = (const float*)d_in[1];
    const float* Wq = (const float*)d_in[2];  const float* bq = (const float*)d_in[3];
    const float* Wk = (const float*)d_in[4];  const float* bk = (const float*)d_in[5];
    const float* Wv = (const float*)d_in[6];  const float* bv = (const float*)d_in[7];
    const float* Wo = (const float*)d_in[8];  const float* bo = (const float*)d_in[9];
    const float* W1 = (const float*)d_in[10]; const float* b1 = (const float*)d_in[11];
    const float* W2 = (const float*)d_in[12]; const float* b2 = (const float*)d_in[13];
    const float* g_img = (const float*)d_in[14]; const float* b_img = (const float*)d_in[15];
    const float* g_hid = (const float*)d_in[16]; const float* b_hid = (const float*)d_in[17];
    const float* g_ffn = (const float*)d_in[18]; const float* b_ffn = (const float*)d_in[19];

    char* ws = (char*)d_ws;
    size_t off = 0;
    auto alloc = [&](size_t bytes) -> void* {
        void* p = ws + off;
        off += (bytes + 255) & ~(size_t)255;
        return p;
    };
    const size_t MQ = (size_t)BATCH * NQ;    // 4096
    const size_t MK = (size_t)BATCH * NKV;   // 36864
    __bf16* xn     = (__bf16*)alloc(MK * IMG_DIM * 2);
    __bf16* hn     = (__bf16*)alloc(MQ * H_DIM * 2);
    __bf16* WqT    = (__bf16*)alloc((size_t)H_DIM * H_DIM * 2);
    __bf16* WkT    = (__bf16*)alloc((size_t)H_DIM * IMG_DIM * 2);
    __bf16* WvT    = (__bf16*)alloc((size_t)H_DIM * IMG_DIM * 2);
    __bf16* WoT    = (__bf16*)alloc((size_t)H_DIM * H_DIM * 2);
    __bf16* W1T    = (__bf16*)alloc((size_t)INTER * H_DIM * 2);
    __bf16* W2T    = (__bf16*)alloc((size_t)H_DIM * INTER * 2);
    __bf16* Qh     = (__bf16*)alloc(MQ * H_DIM * 2);          // [b][h][q][64], pre-scaled
    __bf16* Kh     = (__bf16*)alloc(MK * H_DIM * 2);          // [b][h][key][64]
    __bf16* VtH    = (__bf16*)alloc(MK * H_DIM * 2);          // [b][h][64][key]
    __bf16* attnO  = (__bf16*)alloc(MQ * H_DIM * 2);
    float*  out1   = (float*) alloc(MQ * H_DIM * 4);          // res2 (fp32)
    __bf16* ffnIn  = (__bf16*)alloc(MQ * H_DIM * 2);
    __bf16* ffnMid = (__bf16*)alloc(MQ * INTER * 2);

    // 1) Weights -> bf16, transposed to [N][K]
    auto wt = [&](const float* W, __bf16* Wt, int Kd, int Nd) {
        int total = Kd * Nd;
        transpose_convert_kernel<<<(total + 255) / 256, 256, 0, stream>>>(W, Wt, Kd, Nd);
    };
    wt(Wq, WqT, H_DIM, H_DIM);
    wt(Wk, WkT, IMG_DIM, H_DIM);
    wt(Wv, WvT, IMG_DIM, H_DIM);
    wt(Wo, WoT, H_DIM, H_DIM);
    wt(W1, W1T, H_DIM, INTER);
    wt(W2, W2T, INTER, H_DIM);

    // 2) LayerNorms -> bf16 activations
    ln_kernel<<<(int)MK, 256, 0, stream>>>(x, g_img, b_img, xn, IMG_DIM);
    ln_kernel<<<(int)MQ, 256, 0, stream>>>(hidden, g_hid, b_hid, hn, H_DIM);

    // 3) Projections (WMMA GEMMs), block tile 128x128
    gemm_bf16_kernel<<<dim3(36864 / 128, 768 / 128), 256, 0, stream>>>(
        xn, WkT, bk, Kh, nullptr, nullptr, 36864, H_DIM, IMG_DIM, 1.f, 0, /*mode*/1, NKV);
    gemm_bf16_kernel<<<dim3(36864 / 128, 768 / 128), 256, 0, stream>>>(
        xn, WvT, bv, VtH, nullptr, nullptr, 36864, H_DIM, IMG_DIM, 1.f, 0, /*mode*/2, NKV);
    gemm_bf16_kernel<<<dim3(4096 / 128, 768 / 128), 256, 0, stream>>>(
        hn, WqT, bq, Qh, nullptr, nullptr, 4096, H_DIM, H_DIM, QK_SCALE, 0, /*mode*/1, NQ);

    // 4) Flash attention: 3072 wave-tasks, 4 waves/block
    attention_kernel<<<(BATCH * HEADS * (NQ / 16)) / 4, 128, 0, stream>>>(Qh, Kh, VtH, attnO);

    // 5) Output projection + residual (fp32 res2)
    gemm_bf16_kernel<<<dim3(4096 / 128, 768 / 128), 256, 0, stream>>>(
        attnO, WoT, bo, nullptr, out1, hidden, 4096, H_DIM, H_DIM, 1.f, 0, /*mode*/0, 0);

    // 6) FFN: LN -> GEMM+GELU -> GEMM + residual -> d_out (fp32)
    ln_kernel<<<(int)MQ, 256, 0, stream>>>(out1, g_ffn, b_ffn, ffnIn, H_DIM);
    gemm_bf16_kernel<<<dim3(4096 / 128, 3072 / 128), 256, 0, stream>>>(
        ffnIn, W1T, b1, ffnMid, nullptr, nullptr, 4096, INTER, H_DIM, 1.f, /*gelu*/1, 0, 0);
    gemm_bf16_kernel<<<dim3(4096 / 128, 768 / 128), 256, 0, stream>>>(
        ffnMid, W2T, b2, nullptr, (float*)d_out, out1, 4096, H_DIM, INTER, 1.f, 0, 0, 0);
}
